// GinkaRNNModel_3564822855939
// MI455X (gfx1250) — matrix operand order repaired and checked by reading, not compiled, not run on hardware
//
#include <hip/hip_runtime.h>
#include <cstdint>

typedef __bf16 bf16_t;
typedef __attribute__((ext_vector_type(16))) __bf16 v16bf;
typedef __attribute__((ext_vector_type(8)))  float  v8f;

// typed pointers for the gfx1250 async load-to-LDS builtin:
//   (int4 addrspace(1)* gaddr, int4 addrspace(3)* lds, imm offset, imm cpol)
typedef int gk_int4 __attribute__((vector_size(16)));
typedef gk_int4 __attribute__((address_space(1)))* gk_g4p;
typedef gk_int4 __attribute__((address_space(3)))* gk_l4p;

#if defined(__has_builtin)
# if __has_builtin(__builtin_amdgcn_global_load_async_to_lds_b128)
#  define GK_HAVE_ASYNC 1
# endif
#endif

#define GELU_C 0.70710678118654752f
static __device__ __forceinline__ float gelu_erf(float x) {
  return 0.5f * x * (1.0f + erff(x * GELU_C));
}
static __device__ __forceinline__ float act_apply(float x, int act) {
  if (act == 1) return fmaxf(x, 0.0f);
  if (act == 2) return gelu_erf(x);
  return x;
}

// ---------------- fragment loaders (wave32 CDNA5 WMMA layouts) ----------------
// A matrix 16x32 bf16, row-major source [rows, K]:
//   lane m = l&15 is the M row; halves 0..7 -> K = kk + hi*8 + i,
//   halves 8..15 -> K = kk + hi*8 + 16 + (i-8)   (hi = l>>4)
static __device__ __forceinline__ v16bf load_fragA(const bf16_t* __restrict__ base,
                                                   int row, int K, int kk, int lane) {
  int hi = lane >> 4;
  const bf16_t* p = base + (size_t)row * K + kk + hi * 8;
  v16bf f;
  if (kk + 32 <= K) {
    union { v16bf v; uint4 q[2]; } u;
    u.q[0] = *(const uint4*)(p);
    u.q[1] = *(const uint4*)(p + 16);
    f = u.v;
  } else {
#pragma unroll
    for (int i = 0; i < 16; ++i) {
      int k = kk + hi * 8 + i + (i >= 8 ? 8 : 0);
      f[i] = (k < K) ? base[(size_t)row * K + k] : (bf16_t)0.0f;
    }
  }
  return f;
}
// A fragment from an LDS-staged tile; rowp points at the 32-element row slice.
static __device__ __forceinline__ v16bf load_fragA_lds(const bf16_t* rowp, int lane) {
  int hi = lane >> 4;
  union { v16bf v; uint4 q[2]; } u;
  u.q[0] = *(const uint4*)(rowp + hi * 8);
  u.q[1] = *(const uint4*)(rowp + hi * 8 + 16);
  return u.v;
}
// B matrix 32x16 bf16 from row-major weight [N, K] (column n = weight row n):
//   halves 0..7 -> K = kk + hi*8 + i, halves 8..15 -> K = kk + hi*8 + 16 + (i-8)
static __device__ __forceinline__ v16bf load_fragB(const bf16_t* __restrict__ Wt,
                                                   int rown, int K, int kk, int lane) {
  int hi = lane >> 4;
  const bf16_t* p = Wt + (size_t)rown * K + kk + hi * 8;
  v16bf f;
  if (kk + 32 <= K) {
    union { v16bf v; uint4 q[2]; } u;
    u.q[0] = *(const uint4*)(p);
    u.q[1] = *(const uint4*)(p + 16);
    f = u.v;
  } else {
#pragma unroll
    for (int i = 0; i < 16; ++i) {
      int k = kk + hi * 8 + i + (i >= 8 ? 8 : 0);
      f[i] = (k < K) ? Wt[(size_t)rown * K + k] : (bf16_t)0.0f;
    }
  }
  return f;
}

// ---------------- async A-tile staging: 128 rows x 32 K bf16 = 8 KB ----------
// thread tid covers two 16B segments; 4 consecutive lanes cover one 64B row.
static __device__ __forceinline__ void stage_atile(bf16_t* dst, const bf16_t* __restrict__ A,
                                                   int blockM, int M, int K, int kk, int tid) {
#pragma unroll
  for (int c = 0; c < 2; ++c) {
    int idx = tid + (c << 8);
    int row = idx >> 2, seg = idx & 3;
    int ar = min(blockM + row, M - 1);
    const bf16_t* gp = A + (size_t)ar * K + kk + seg * 8;
    bf16_t* lp = dst + row * 32 + seg * 8;
#if defined(GK_HAVE_ASYNC)
    __builtin_amdgcn_global_load_async_to_lds_b128(
        (gk_g4p)(uintptr_t)gp, (gk_l4p)(uint32_t)(uintptr_t)lp, 0, 0);
#else
    *(uint4*)lp = *(const uint4*)gp;
#endif
  }
}
static __device__ __forceinline__ void async_wait_stage(int more) {
#if defined(GK_HAVE_ASYNC)
# if __has_builtin(__builtin_amdgcn_s_wait_asynccnt)
  if (more) __builtin_amdgcn_s_wait_asynccnt(2);
  else      __builtin_amdgcn_s_wait_asynccnt(0);
# else
  if (more) asm volatile("s_wait_asynccnt 0x2" ::: "memory");
  else      asm volatile("s_wait_asynccnt 0x0" ::: "memory");
# endif
#else
  (void)more;
#endif
}

// ---------------- generic WMMA GEMM:  C[M,N] = act(A[M,K] @ W[N,K]^T + bias) --
// block = 256 threads = 8 waves arranged 4(M) x 2(N); block tile 128x64;
// each wave computes a 32x32 tile = 4 v_wmma_f32_16x16x32_bf16 per K-step.
// A tiles are double-buffered through LDS via async load-to-LDS (coalesced);
// W rows (L2-resident) are loaded directly as fragments, with prefetch.
template <typename CT>
__global__ __launch_bounds__(256) void gemm_wmma(
    const bf16_t* __restrict__ A, const bf16_t* __restrict__ Wt,
    const float* __restrict__ bias, CT* __restrict__ C,
    int M, int N, int K, int act)
{
  __shared__ __align__(16) bf16_t atile[2][128 * 32];
  int tid = threadIdx.x;
  int lane = tid & 31;
  int wave = tid >> 5;
  int wm = wave & 3, wn = wave >> 2;
  int m0 = blockIdx.y * 128 + wm * 32;
  int n0 = blockIdx.x * 64 + wn * 32;

  v8f acc[2][2] = {};
  int rb[2];
#pragma unroll
  for (int s = 0; s < 2; ++s)
    rb[s] = min(n0 + s * 16 + (lane & 15), N - 1);

  if ((K & 31) == 0) {
    // ---- LDS-staged, double-buffered path ----
    int lr0 = (wm * 32 + 0  + (lane & 15)) * 32;
    int lr1 = (wm * 32 + 16 + (lane & 15)) * 32;
    stage_atile(atile[0], A, blockIdx.y * 128, M, K, 0, tid);
    for (int kk = 0; kk < K; kk += 32) {
      int cur = (kk >> 5) & 1;
      int more = (kk + 32 < K);
      if (more) {
        stage_atile(atile[cur ^ 1], A, blockIdx.y * 128, M, K, kk + 32, tid);
        __builtin_prefetch(Wt + (size_t)rb[0] * K + kk + 32, 0, 1);
        __builtin_prefetch(Wt + (size_t)rb[1] * K + kk + 32, 0, 1);
      }
      async_wait_stage(more);
      __syncthreads();
      v16bf b0 = load_fragB(Wt, rb[0], K, kk, lane);
      v16bf b1 = load_fragB(Wt, rb[1], K, kk, lane);
      v16bf a0 = load_fragA_lds(&atile[cur][lr0], lane);
      v16bf a1 = load_fragA_lds(&atile[cur][lr1], lane);
      acc[0][0] = __builtin_amdgcn_wmma_f32_16x16x32_bf16(false, a0, false, b0, (short)0, acc[0][0], false, false);
      acc[0][1] = __builtin_amdgcn_wmma_f32_16x16x32_bf16(false, a0, false, b1, (short)0, acc[0][1], false, false);
      acc[1][0] = __builtin_amdgcn_wmma_f32_16x16x32_bf16(false, a1, false, b0, (short)0, acc[1][0], false, false);
      acc[1][1] = __builtin_amdgcn_wmma_f32_16x16x32_bf16(false, a1, false, b1, (short)0, acc[1][1], false, false);
      __syncthreads();  // all waves done with atile[cur] before it is re-staged
    }
  } else {
    // ---- direct path (handles K not multiple of 32, e.g. K=16) ----
    int ra[2];
#pragma unroll
    for (int s = 0; s < 2; ++s)
      ra[s] = min(m0 + s * 16 + (lane & 15), M - 1);
    for (int kk = 0; kk < K; kk += 32) {
      v16bf a0 = load_fragA(A, ra[0], K, kk, lane);
      v16bf a1 = load_fragA(A, ra[1], K, kk, lane);
      v16bf b0 = load_fragB(Wt, rb[0], K, kk, lane);
      v16bf b1 = load_fragB(Wt, rb[1], K, kk, lane);
      acc[0][0] = __builtin_amdgcn_wmma_f32_16x16x32_bf16(false, a0, false, b0, (short)0, acc[0][0], false, false);
      acc[0][1] = __builtin_amdgcn_wmma_f32_16x16x32_bf16(false, a0, false, b1, (short)0, acc[0][1], false, false);
      acc[1][0] = __builtin_amdgcn_wmma_f32_16x16x32_bf16(false, a1, false, b0, (short)0, acc[1][0], false, false);
      acc[1][1] = __builtin_amdgcn_wmma_f32_16x16x32_bf16(false, a1, false, b1, (short)0, acc[1][1], false, false);
    }
  }

#pragma unroll
  for (int si = 0; si < 2; ++si) {
#pragma unroll
    for (int sj = 0; sj < 2; ++sj) {
      int n = n0 + sj * 16 + (lane & 15);
      if (n >= N) continue;
      float bv = bias ? bias[n] : 0.0f;
#pragma unroll
      for (int j = 0; j < 8; ++j) {
        int m = m0 + si * 16 + (lane >> 4) * 8 + j;
        if (m < M) {
          float v = act_apply(acc[si][sj][j] + bv, act);
          C[(size_t)m * N + n] = (CT)v;
        }
      }
    }
  }
}

// ---------------- f32 -> bf16 converter ----------------
__global__ void cvt_f32_bf16(const float* __restrict__ in, bf16_t* __restrict__ out, int n) {
  int i = blockIdx.x * 256 + threadIdx.x;
  if (i < n) out[i] = (bf16_t)in[i];
}

// ---------------- LayerNorm (optional residual add, optional fused GELU) -----
__global__ __launch_bounds__(256) void ln_kernel(
    const bf16_t* __restrict__ X, const bf16_t* __restrict__ Rz,
    const float* __restrict__ g, const float* __restrict__ b,
    bf16_t* __restrict__ out, int Dm, int dogelu)
{
  __shared__ float redA[8];
  __shared__ float redB[8];
  __shared__ float stats[2];
  size_t row = blockIdx.x;
  const bf16_t* xr = X + row * Dm;
  const bf16_t* rr = Rz ? Rz + row * Dm : nullptr;
  int cnt = Dm >> 8;                 // Dm in {256,512,1024}
  float vals[4];
  float s = 0.0f;
  for (int c = 0; c < cnt; ++c) {
    int d = (c << 8) + threadIdx.x;
    float v = (float)xr[d] + (rr ? (float)rr[d] : 0.0f);
    vals[c] = v; s += v;
  }
  for (int o = 16; o > 0; o >>= 1) s += __shfl_xor(s, o, 32);
  if ((threadIdx.x & 31) == 0) redA[threadIdx.x >> 5] = s;
  __syncthreads();
  if (threadIdx.x == 0) {
    float t2 = 0.0f;
    for (int k = 0; k < 8; ++k) t2 += redA[k];
    stats[0] = t2 / (float)Dm;
  }
  __syncthreads();
  float mean = stats[0];
  float vs = 0.0f;
  for (int c = 0; c < cnt; ++c) { float dv = vals[c] - mean; vs += dv * dv; }
  for (int o = 16; o > 0; o >>= 1) vs += __shfl_xor(vs, o, 32);
  if ((threadIdx.x & 31) == 0) redB[threadIdx.x >> 5] = vs;
  __syncthreads();
  if (threadIdx.x == 0) {
    float t2 = 0.0f;
    for (int k = 0; k < 8; ++k) t2 += redB[k];
    stats[1] = rsqrtf(t2 / (float)Dm + 1e-5f);
  }
  __syncthreads();
  float inv = stats[1];
  for (int c = 0; c < cnt; ++c) {
    int d = (c << 8) + threadIdx.x;
    float y = (vals[c] - mean) * inv * g[d] + b[d];
    if (dogelu) y = gelu_erf(y);
    out[row * Dm + d] = (bf16_t)y;
  }
}

// ---------------- fused patch encoder ----------------------------------------
// one block per patch p = t*128 + b:
//  conv1 (one-hot gather, VALU) -> LDS, conv2 as WMMA GEMM (M=512,K=2304,N=9),
//  BN+GELU epilogues, then 4608->256 FC, all in one kernel.
__global__ __launch_bounds__(256) void patch_encoder(
    const int* __restrict__ tmap,
    const float* __restrict__ pc1w, const float* __restrict__ pc1b,
    const float* __restrict__ bn1g, const float* __restrict__ bn1b,
    const bf16_t* __restrict__ pc2wb, const float* __restrict__ pc2b,
    const float* __restrict__ bn2g, const float* __restrict__ bn2b,
    const bf16_t* __restrict__ pfwb, const float* __restrict__ pfb,
    bf16_t* __restrict__ patchf)
{
  __shared__ int   res_arr[5][5];
  __shared__ float mask_arr[5][5];
  __shared__ bf16_t h1l[256 * 25];
  __shared__ __align__(16) bf16_t bcol[8 * 32 * 16];
  __shared__ float hh2l[4608];

  int p = blockIdx.x;
  int t = p >> 7, b = p & 127;
  int y = t / 13, x = t % 13;
  int tid = threadIdx.x;
  int lane = tid & 31, wave = tid >> 5;

  if (tid < 25) {
    int py = tid / 5, px = tid % 5;
    int gy = y - 4 + py, gx = x - 2 + px;
    bool ok = (gy >= 0) && (gx >= 0) && (gx < 13) && !(py == 4 && px >= 2);
    res_arr[py][px]  = ok ? tmap[b * 169 + gy * 13 + gx] : 0;
    mask_arr[py][px] = ok ? 1.0f : 0.0f;
  }
  __syncthreads();

  // conv1 + BN + GELU; channels 0..31 one-hot of transposed patch, ch 32 = mask
  {
    int oc = tid;
    float s1 = bn1g[oc] * rsqrtf(1.0f + 1e-5f);
    float sh = bn1b[oc];
    float bz = pc1b[oc];
    const float* wrow = pc1w + (size_t)oc * 33 * 9;
    for (int pos = 0; pos < 25; ++pos) {
      int i = pos / 5, j = pos % 5;
      float acc = bz;
#pragma unroll
      for (int ki = 0; ki < 3; ++ki) {
        int ii = i + ki - 1;
        if (ii < 0 || ii >= 5) continue;
#pragma unroll
        for (int kj = 0; kj < 3; ++kj) {
          int jj = j + kj - 1;
          if (jj < 0 || jj >= 5) continue;
          int c = res_arr[jj][ii];  // H/W-swapped one-hot
          acc += wrow[c * 9 + ki * 3 + kj] + mask_arr[ii][jj] * wrow[32 * 9 + ki * 3 + kj];
        }
      }
      float v = acc * s1 + sh;
      v = gelu_erf(v);
      h1l[oc * 25 + pos] = (bf16_t)v;
    }
  }

  // conv2 as WMMA: A = W2[512,2304] bf16 (L2-resident), B = im2col from LDS
  v8f acc[4] = {};
  for (int kc = 0; kc < 72; kc += 8) {
    __syncthreads();  // previous chunk consumed
    for (int idx = tid; idx < 4096; idx += 256) {
      int ks2 = idx >> 9;
      int ln2 = (idx >> 4) & 31;
      int ii  = idx & 15;
      int n   = ln2 & 15;
      int k   = (kc + ks2) * 32 + (ln2 >> 4) * 8 + ii + (ii >= 8 ? 8 : 0);
      bf16_t v = (bf16_t)0.0f;
      if (n < 9) {
        int c = k / 9, r = k % 9;
        int ki = r / 3, kj = r % 3;
        int oy = n / 3, ox = n % 3;
        v = h1l[c * 25 + (oy + ki) * 5 + (ox + kj)];
      }
      bcol[idx] = v;
    }
    __syncthreads();
    for (int ks2 = 0; ks2 < 8; ++ks2) {
      v16bf bf;
      {
        union { v16bf v; uint4 q[2]; } u;
        const bf16_t* bp = &bcol[(ks2 * 32 + lane) * 16];
        u.q[0] = *(const uint4*)(bp);
        u.q[1] = *(const uint4*)(bp + 8);
        bf = u.v;
      }
      int kk = (kc + ks2) * 32;
#pragma unroll
      for (int mt = 0; mt < 4; ++mt) {
        int rowa = (wave + mt * 8) * 16 + (lane & 15);
        v16bf af = load_fragA(pc2wb, rowa, 2304, kk, lane);
        acc[mt] = __builtin_amdgcn_wmma_f32_16x16x32_bf16(false, af, false, bf, (short)0, acc[mt], false, false);
      }
    }
  }
  __syncthreads();

  // bias + BN + GELU epilogue -> hh2l[od*9 + n]
  {
    float s2c = rsqrtf(1.0f + 1e-5f);
    int n = lane & 15;
    if (n < 9) {
#pragma unroll
      for (int mt = 0; mt < 4; ++mt) {
#pragma unroll
        for (int j = 0; j < 8; ++j) {
          int od = (wave + mt * 8) * 16 + (lane >> 4) * 8 + j;
          float v = acc[mt][j] + pc2b[od];
          v = v * (bn2g[od] * s2c) + bn2b[od];
          hh2l[od * 9 + n] = gelu_erf(v);
        }
      }
    }
  }
  __syncthreads();

  // FC 4608 -> 256
  {
    int d = tid;
    float a2 = pfb[d];
    const bf16_t* wr = pfwb + (size_t)d * 4608;
    for (int k = 0; k < 4608; k += 4) {
      a2 += (float)wr[k]     * hh2l[k]
          + (float)wr[k + 1] * hh2l[k + 1]
          + (float)wr[k + 2] * hh2l[k + 2]
          + (float)wr[k + 3] * hh2l[k + 3];
    }
    patchf[(size_t)p * 256 + d] = (bf16_t)a2;
  }
}

// ---------------- token assembly ----------------------------------------------
__global__ __launch_bounds__(256) void token_build(
    const int* __restrict__ tmap, const float* __restrict__ tile_emb,
    const float* __restrict__ row_emb, const float* __restrict__ col_emb,
    const float* __restrict__ condf, const bf16_t* __restrict__ patchf,
    bf16_t* __restrict__ tok)
{
  int p = blockIdx.x;
  int t = p >> 7, b = p & 127;
  int d = threadIdx.x;
  int prev = (t == 0) ? 31 : tmap[b * 169 + (t - 1)];
  int xs = t % 13, ys = t / 13;
  size_t base = (size_t)p * 5 * 256;
  tok[base + 0 * 256 + d] = (bf16_t)tile_emb[prev * 256 + d];
  tok[base + 1 * 256 + d] = (bf16_t)condf[b * 256 + d];
  tok[base + 2 * 256 + d] = (bf16_t)row_emb[xs * 256 + d];
  tok[base + 3 * 256 + d] = (bf16_t)col_emb[ys * 256 + d];
  tok[base + 4 * 256 + d] = patchf[(size_t)p * 256 + d];
}

// ---------------- attention (S=5, 2 heads, hd=128) ----------------------------
__global__ __launch_bounds__(256) void attn_kernel(const bf16_t* __restrict__ qkv,
                                                   bf16_t* __restrict__ out)
{
  __shared__ float sc[2][5][5];
  __shared__ float at[2][5][5];
  int p = blockIdx.x;
  const bf16_t* base = qkv + (size_t)p * 5 * 768;
  int tid = threadIdx.x;
  if (tid < 50) {
    int h = tid / 25, r = tid % 25, i = r / 5, kx = r % 5;
    const bf16_t* q  = base + i * 768 + h * 128;
    const bf16_t* kk = base + kx * 768 + 256 + h * 128;
    float sdot = 0.0f;
    for (int d = 0; d < 128; ++d) sdot += (float)q[d] * (float)kk[d];
    sc[h][i][kx] = sdot * 0.08838834764831845f;  // 1/sqrt(128)
  }
  __syncthreads();
  if (tid < 10) {
    int h = tid / 5, i = tid % 5;
    float mx = -1e30f;
    for (int kx = 0; kx < 5; ++kx) mx = fmaxf(mx, sc[h][i][kx]);
    float ssum = 0.0f, e[5];
    for (int kx = 0; kx < 5; ++kx) { e[kx] = __expf(sc[h][i][kx] - mx); ssum += e[kx]; }
    for (int kx = 0; kx < 5; ++kx) at[h][i][kx] = e[kx] / ssum;
  }
  __syncthreads();
  int h = tid >> 7, d = tid & 127;
  for (int i = 0; i < 5; ++i) {
    float o = 0.0f;
    for (int kx = 0; kx < 5; ++kx)
      o += at[h][i][kx] * (float)base[kx * 768 + 512 + h * 128 + d];
    out[(size_t)p * 5 * 256 + i * 256 + h * 128 + d] = (bf16_t)o;
  }
}

// ---------------- misc small kernels ------------------------------------------
__global__ void gather_feat(const bf16_t* __restrict__ tok, bf16_t* __restrict__ featb) {
  int idx = blockIdx.x * 256 + threadIdx.x;
  if (idx >= 21632 * 256) return;
  int p = idx >> 8, d = idx & 255;
  featb[idx] = tok[(size_t)p * 5 * 256 + d];
}

__global__ void zero_h(float* __restrict__ h, bf16_t* __restrict__ hb, int n) {
  int i = blockIdx.x * 256 + threadIdx.x;
  if (i < n) { h[i] = 0.0f; hb[i] = (bf16_t)0.0f; }
}

__global__ void gru_gate(const float* __restrict__ gi, const float* __restrict__ gh,
                         float* __restrict__ h, bf16_t* __restrict__ hb,
                         bf16_t* __restrict__ hiddens, int t)
{
  int idx = blockIdx.x * 256 + threadIdx.x;
  if (idx >= 128 * 512) return;
  int b = idx >> 9, u = idx & 511;
  const float* gir = gi + ((size_t)(t * 128 + b)) * 1536;
  const float* ghr = gh + (size_t)b * 1536;
  float r = 1.0f / (1.0f + __expf(-(gir[u] + ghr[u])));
  float z = 1.0f / (1.0f + __expf(-(gir[512 + u] + ghr[512 + u])));
  float n = tanhf(gir[1024 + u] + r * ghr[1024 + u]);
  float hv = (1.0f - z) * n + z * h[idx];
  h[idx] = hv;
  hb[idx] = (bf16_t)hv;
  hiddens[(size_t)t * 128 * 512 + idx] = (bf16_t)hv;
}

__global__ void out_kernel(const float* __restrict__ logits, float* __restrict__ outp,
                           int* __restrict__ mp)
{
  __shared__ float sv[32];
  int p = blockIdx.x;
  int t = p >> 7, b = p & 127;
  int c = threadIdx.x;
  float v = logits[(size_t)p * 32 + c];
  outp[((size_t)b * 32 + c) * 169 + t] = v;
  sv[c] = v;
  __syncthreads();
  if (c == 0) {
    int am = 0; float mx = sv[0];
    for (int k = 1; k < 32; ++k) if (sv[k] > mx) { mx = sv[k]; am = k; }
    mp[b * 169 + t] = am;
  }
}

// ---------------- host orchestration -------------------------------------------
extern "C" void kernel_launch(void* const* d_in, const int* in_sizes, int n_in,
                              void* d_out, int out_size, void* d_ws, size_t ws_size,
                              hipStream_t stream)
{
  (void)in_sizes; (void)n_in; (void)out_size; (void)ws_size;
  // ---- inputs (setup_inputs dict insertion order) ----
  const float* val_cond = (const float*)d_in[0];
  const int*   tmap     = (const int*)d_in[1];
  int ix = 2;
  const float* cw1   = (const float*)d_in[ix++];  // [1024,16]
  const float* cb1   = (const float*)d_in[ix++];
  const float* clg   = (const float*)d_in[ix++];
  const float* clb   = (const float*)d_in[ix++];
  const float* cw2   = (const float*)d_in[ix++];  // [256,1024]
  const float* cb2   = (const float*)d_in[ix++];
  const float* tile_emb = (const float*)d_in[ix++];
  const float* row_emb  = (const float*)d_in[ix++];
  const float* col_emb  = (const float*)d_in[ix++];
  const float* pc1w  = (const float*)d_in[ix++];  // [256,33,3,3]
  const float* pc1b  = (const float*)d_in[ix++];
  const float* pbn1g = (const float*)d_in[ix++];
  const float* pbn1b = (const float*)d_in[ix++];
  const float* pc2w  = (const float*)d_in[ix++];  // [512,256,3,3]
  const float* pc2b  = (const float*)d_in[ix++];
  const float* pbn2g = (const float*)d_in[ix++];
  const float* pbn2b = (const float*)d_in[ix++];
  const float* pfw   = (const float*)d_in[ix++];  // [256,4608]
  const float* pfb   = (const float*)d_in[ix++];
  const float *e_wi[4], *e_bi[4], *e_wo[4], *e_bo[4], *e_g1[4], *e_b1[4],
              *e_w1[4], *e_bb1[4], *e_w2[4], *e_bb2[4], *e_g2[4], *e_b2[4];
  for (int l = 0; l < 4; ++l) {
    e_wi[l]  = (const float*)d_in[ix++]; e_bi[l]  = (const float*)d_in[ix++];
    e_wo[l]  = (const float*)d_in[ix++]; e_bo[l]  = (const float*)d_in[ix++];
    e_g1[l]  = (const float*)d_in[ix++]; e_b1[l]  = (const float*)d_in[ix++];
    e_w1[l]  = (const float*)d_in[ix++]; e_bb1[l] = (const float*)d_in[ix++];
    e_w2[l]  = (const float*)d_in[ix++]; e_bb2[l] = (const float*)d_in[ix++];
    e_g2[l]  = (const float*)d_in[ix++]; e_b2[l]  = (const float*)d_in[ix++];
  }
  const float* gwih = (const float*)d_in[ix++];  // [1536,256]
  const float* gwhh = (const float*)d_in[ix++];  // [1536,512]
  const float* gbih = (const float*)d_in[ix++];
  const float* gbhh = (const float*)d_in[ix++];
  const float* fw1  = (const float*)d_in[ix++];  // [512,512]
  const float* fb1  = (const float*)d_in[ix++];
  const float* flg  = (const float*)d_in[ix++];
  const float* flb  = (const float*)d_in[ix++];
  const float* fw2  = (const float*)d_in[ix++];  // [32,512]
  const float* fb2  = (const float*)d_in[ix++];

  // ---- workspace carve (bump allocator) ----
  char* Wb = (char*)d_ws;
  size_t off = 0;
  auto AL = [&](size_t bytes) -> void* {
    void* p = (void*)(Wb + off);
    off = (off + bytes + 255) & ~(size_t)255;
    return p;
  };
  bf16_t* vcb     = (bf16_t*)AL(128 * 16 * 2);
  bf16_t* b_cw1   = (bf16_t*)AL(1024 * 16 * 2);
  bf16_t* b_cw2   = (bf16_t*)AL(256 * 1024 * 2);
  bf16_t* b_pc2w  = (bf16_t*)AL(512 * 2304 * 2);
  bf16_t* b_pfw   = (bf16_t*)AL(256 * 4608 * 2);
  bf16_t *b_wi[4], *b_wo[4], *b_w1[4], *b_w2[4];
  for (int l = 0; l < 4; ++l) {
    b_wi[l] = (bf16_t*)AL(768 * 256 * 2);
    b_wo[l] = (bf16_t*)AL(256 * 256 * 2);
    b_w1[l] = (bf16_t*)AL(512 * 256 * 2);
    b_w2[l] = (bf16_t*)AL(256 * 512 * 2);
  }
  bf16_t* b_gwih = (bf16_t*)AL(1536 * 256 * 2);
  bf16_t* b_gwhh = (bf16_t*)AL(1536 * 512 * 2);
  bf16_t* b_fw1  = (bf16_t*)AL(512 * 512 * 2);
  bf16_t* b_fw2  = (bf16_t*)AL(32 * 512 * 2);

  bf16_t* vhpre  = (bf16_t*)AL(128 * 1024 * 2);
  bf16_t* vhb    = (bf16_t*)AL(128 * 1024 * 2);
  float*  condf  = (float*)AL(128 * 256 * 4);
  bf16_t* patchf = (bf16_t*)AL((size_t)21632 * 256 * 2);
  bf16_t* tok    = (bf16_t*)AL((size_t)108160 * 256 * 2);
  void*   qkv_gi = AL((size_t)108160 * 768 * 2);          // later reused as f32 gi [21632,1536]
  bf16_t* bigbuf = (bf16_t*)AL((size_t)108160 * 512 * 2); // attn-concat / FF / head temporaries
  bf16_t* projb  = (bf16_t*)AL((size_t)108160 * 256 * 2);
  bf16_t* featb  = (bf16_t*)AL((size_t)21632 * 256 * 2);
  bf16_t* hiddb  = (bf16_t*)AL((size_t)21632 * 512 * 2);
  float*  hf     = (float*)AL(128 * 512 * 4);
  bf16_t* hbb    = (bf16_t*)AL(128 * 512 * 2);
  float*  ghf    = (float*)AL(128 * 1536 * 4);
  float*  logitsf = (float*)AL((size_t)21632 * 32 * 4);

  bf16_t* qkvb = (bf16_t*)qkv_gi;
  float*  gif  = (float*)qkv_gi;
  bf16_t* abuf = bigbuf;               // [108160,256]
  bf16_t* ffb  = bigbuf;               // [108160,512]
  bf16_t* hh1b = bigbuf;               // [21632,512]
  bf16_t* hhb2 = bigbuf + (size_t)21632 * 512;

  auto CVT = [&](const float* src, bf16_t* dst, int n) {
    cvt_f32_bf16<<<(n + 255) / 256, 256, 0, stream>>>(src, dst, n);
  };
  auto GEMM_B = [&](const bf16_t* A, const bf16_t* Wt, const float* bias, bf16_t* C,
                    int M, int N, int K, int act) {
    dim3 g((N + 63) / 64, (M + 127) / 128);
    gemm_wmma<bf16_t><<<g, 256, 0, stream>>>(A, Wt, bias, C, M, N, K, act);
  };
  auto GEMM_F = [&](const bf16_t* A, const bf16_t* Wt, const float* bias, float* C,
                    int M, int N, int K, int act) {
    dim3 g((N + 63) / 64, (M + 127) / 128);
    gemm_wmma<float><<<g, 256, 0, stream>>>(A, Wt, bias, C, M, N, K, act);
  };

  // ---- weight / input conversions to bf16 ----
  CVT(val_cond, vcb, 128 * 16);
  CVT(cw1, b_cw1, 1024 * 16);
  CVT(cw2, b_cw2, 256 * 1024);
  CVT(pc2w, b_pc2w, 512 * 2304);
  CVT(pfw, b_pfw, 256 * 4608);
  for (int l = 0; l < 4; ++l) {
    CVT(e_wi[l], b_wi[l], 768 * 256);
    CVT(e_wo[l], b_wo[l], 256 * 256);
    CVT(e_w1[l], b_w1[l], 512 * 256);
    CVT(e_w2[l], b_w2[l], 256 * 512);
  }
  CVT(gwih, b_gwih, 1536 * 256);
  CVT(gwhh, b_gwhh, 1536 * 512);
  CVT(fw1, b_fw1, 512 * 512);
  CVT(fw2, b_fw2, 32 * 512);

  // ---- condition encoder ----
  GEMM_B(vcb, b_cw1, cb1, vhpre, 128, 1024, 16, 0);
  ln_kernel<<<128, 256, 0, stream>>>(vhpre, nullptr, clg, clb, vhb, 1024, 1);
  GEMM_F(vhb, b_cw2, cb2, condf, 128, 256, 1024, 0);

  // ---- fused patch encoder (conv1 gather + conv2 WMMA + FC) ----
  patch_encoder<<<21632, 256, 0, stream>>>(tmap, pc1w, pc1b, pbn1g, pbn1b,
                                           b_pc2w, pc2b, pbn2g, pbn2b,
                                           b_pfw, pfb, patchf);

  // ---- token assembly ----
  token_build<<<21632, 256, 0, stream>>>(tmap, tile_emb, row_emb, col_emb,
                                         condf, patchf, tok);

  // ---- 4 transformer encoder layers (post-LN) ----
  for (int l = 0; l < 4; ++l) {
    GEMM_B(tok, b_wi[l], e_bi[l], qkvb, 108160, 768, 256, 0);
    attn_kernel<<<21632, 256, 0, stream>>>(qkvb, abuf);
    GEMM_B(abuf, b_wo[l], e_bo[l], projb, 108160, 256, 256, 0);
    ln_kernel<<<108160, 256, 0, stream>>>(tok, projb, e_g1[l], e_b1[l], tok, 256, 0);
    GEMM_B(tok, b_w1[l], e_bb1[l], ffb, 108160, 512, 256, 1);      // ReLU fused
    GEMM_B(ffb, b_w2[l], e_bb2[l], projb, 108160, 256, 512, 0);
    ln_kernel<<<108160, 256, 0, stream>>>(tok, projb, e_g2[l], e_b2[l], tok, 256, 0);
  }

  // ---- take token 0 as feat ----
  gather_feat<<<(21632 * 256 + 255) / 256, 256, 0, stream>>>(tok, featb);

  // ---- GRU: batched input GEMM, then 169-step recurrent scan ----
  GEMM_F(featb, b_gwih, gbih, gif, 21632, 1536, 256, 0);
  zero_h<<<(128 * 512 + 255) / 256, 256, 0, stream>>>(hf, hbb, 128 * 512);
  for (int t = 0; t < 169; ++t) {
    GEMM_F(hbb, b_gwhh, gbhh, ghf, 128, 1536, 512, 0);
    gru_gate<<<(128 * 512 + 255) / 256, 256, 0, stream>>>(gif, ghf, hf, hbb, hiddb, t);
  }

  // ---- output head ----
  GEMM_B(hiddb, b_fw1, fb1, hh1b, 21632, 512, 512, 0);
  ln_kernel<<<21632, 256, 0, stream>>>(hh1b, nullptr, flg, flb, hhb2, 512, 1);
  GEMM_F(hhb2, b_fw2, fb2, logitsf, 21632, 32, 512, 0);

  float* outp = (float*)d_out;
  int*   mp   = (int*)(outp + (size_t)128 * 32 * 169);
  out_kernel<<<21632, 32, 0, stream>>>(logitsf, outp, mp);
}
